// Iter_block_51960514347739
// MI455X (gfx1250) — compile-verified, matrix operand order checked by reading, not run on hardware
//
#include <hip/hip_runtime.h>
#include <hip/hip_bf16.h>

// ---------------- problem constants (match reference) ----------------
#define NNODES 3969      // patch nodes per image (63*63)
#define PHS    63        // patches per side
#define IMG    256
#define IMG2   65536     // IMG*IMG
#define BATCH  4
#define HID    64        // conv hidden channels
#define GH     128       // gcn hidden features
#define PP     64        // patch pixels (8*8)
#define K2     576       // conv2 implicit-GEMM K = 64*3*3

typedef __bf16 bf16_t;
typedef bf16_t v16bf __attribute__((ext_vector_type(16)));
typedef float  v8f   __attribute__((ext_vector_type(8)));
typedef unsigned int u32x4 __attribute__((ext_vector_type(4)));
typedef int          i32x4 __attribute__((ext_vector_type(4)));
typedef int          i32x8 __attribute__((ext_vector_type(8)));

// K index held by element i of a lane's 16x32 (A) / 32x16 (B) bf16 fragment.
// lanes 0-15: K in {0..7} U {16..23}; lanes 16-31: K in {8..15} U {24..31}.
// Runs of 8 consecutive elements -> b128 loads when the K axis is contiguous.
__device__ __forceinline__ int frag_k(int i, int kq) {
    return (i & 7) + (kq << 3) + ((i >> 3) << 4);
}

// ---------------------------------------------------------------------
// Tensor Data Mover: 2-D tile (tile0 contiguous elements x tile1 rows,
// row stride stride0, all in 2-byte elements) from global to LDS.
// D# per CDNA5 ISA.  Lands contiguously in LDS as
// [tile1][tile0].  Builtin arity differs per toolchain (bridge doc).
// ---------------------------------------------------------------------
__device__ __forceinline__ void tdm_load_2d(unsigned lds_off, const void* gptr,
                                            unsigned tile0, unsigned tile1,
                                            unsigned stride0, unsigned tdim1) {
    unsigned long long ga = (unsigned long long)(uintptr_t)gptr;
    u32x4 g0 = { 1u,                                   // count=1, user mode
                 lds_off,                              // lds_addr
                 (unsigned)ga,                         // global_addr[31:0]
                 (unsigned)((ga >> 32) & 0x01FFFFFFu)  // global_addr[56:32]
                     | (2u << 30) };                   // type=2 ("image")
    i32x8 g1 = { (int)(1u << 16),                      // data_size = 1 -> 2 bytes
                 (int)((stride0 & 0xffffu) << 16),     // tensor_dim0[15:0]
                 (int)(((stride0 >> 16) & 0xffffu)     // tensor_dim0[31:16]
                       | ((tdim1 & 0xffffu) << 16)),   // tensor_dim1[15:0]
                 (int)(((tdim1 >> 16) & 0xffffu)       // tensor_dim1[31:16]
                       | ((tile0 & 0xffffu) << 16)),   // tile_dim0
                 (int)(tile1 & 0xffffu),               // tile_dim1 (tile_dim2=0)
                 (int)stride0,                         // tensor_dim0_stride lo32
                 0, 0 };                               // stride hi16, dim1_stride=0
    i32x4 z4 = { 0, 0, 0, 0 };
#if defined(__clang_major__) && (__clang_major__ >= 23)
    i32x8 z8 = { 0, 0, 0, 0, 0, 0, 0, 0 };
    __builtin_amdgcn_tensor_load_to_lds(g0, g1, z4, z4, z8, 0);
#else
    __builtin_amdgcn_tensor_load_to_lds(g0, g1, z4, z4, 0);
#endif
}

// ---------------------------------------------------------------------
// conv1: 1 -> 64 channels, 3x3 SAME, relu.  (small: direct VALU)
// ---------------------------------------------------------------------
__global__ void conv1_relu_kernel(const float* __restrict__ in,
                                  const float* __restrict__ w,    // [64][1][3][3]
                                  const float* __restrict__ bias, // [64]
                                  float* __restrict__ out) {
    int idx = blockIdx.x * blockDim.x + threadIdx.x;   // (b*64+oc)*IMG2 + pix
    int pix = idx & (IMG2 - 1);
    int oc  = (idx >> 16) & 63;
    int b   = idx >> 22;
    int y = pix >> 8, x = pix & 255;
    float s = bias[oc];
#pragma unroll
    for (int ky = 0; ky < 3; ++ky)
#pragma unroll
        for (int kx = 0; kx < 3; ++kx) {
            int yy = y + ky - 1, xx = x + kx - 1;
            if (yy >= 0 && yy < IMG && xx >= 0 && xx < IMG)
                s += w[oc * 9 + ky * 3 + kx] * in[b * IMG2 + yy * IMG + xx];
        }
    out[idx] = s > 0.f ? s : 0.f;
}

// ---------------------------------------------------------------------
// conv2 (64->64, 3x3, relu) as WMMA implicit GEMM, K = (ky*3+kx)*64 + ic.
// LDS layouts chosen so BOTH fragment operands are contiguous in K:
//   wlds[oc][r][ic]    (A: k runs contiguous)          73728 B
//   hlds[ry][cx][ic]   (B: ic runs contiguous)         13056 B
// Block = 256 thr = 8 waves covering 64 oc x 32 pixels of one image row.
// ---------------------------------------------------------------------
__global__ void conv2_relu_wmma_kernel(const float* __restrict__ h1,
                                       const float* __restrict__ w,    // [64][64][3][3]
                                       const float* __restrict__ bias, // [64]
                                       float* __restrict__ h2) {
    extern __shared__ char dynsm[];
    bf16_t* wlds = (bf16_t*)dynsm;          // [64][9][64]
    bf16_t* hlds = wlds + HID * K2;         // [3][34][64]

    int tid  = threadIdx.x;
    int b    = blockIdx.y;
    int yrow = blockIdx.x >> 3;             // image row
    int x0   = (blockIdx.x & 7) << 5;       // first of 32 pixels in the row

    // stage re-ordered bf16 weights (one-time div/mod, outside MMA loop)
    for (int t = tid; t < HID * K2; t += 256) {
        int oc = t / K2, k = t % K2;
        int ic = k / 9, r = k % 9;
        wlds[oc * K2 + r * HID + ic] = (bf16_t)w[t];
    }
    // stage input halo tile: coalesced global reads, transposed LDS writes
    for (int t = tid; t < HID * 3 * 34; t += 256) {
        int ic = t / 102, rem = t % 102, ry = rem / 34, cx = rem % 34;
        int yy = yrow + ry - 1, xx = x0 + cx - 1;
        float v = 0.f;
        if (yy >= 0 && yy < IMG && xx >= 0 && xx < IMG)
            v = h1[((size_t)(b * HID + ic) << 16) + yy * IMG + xx];
        hlds[((ry * 34 + cx) << 6) + ic] = (bf16_t)v;
    }
    __syncthreads();

    int lane = tid & 31, laneLo = lane & 15, kq = lane >> 4;
    int wave = tid >> 5, mtile = wave & 3, psub = wave >> 2;
    int lc = psub * 16 + laneLo;            // this lane's pixel (B column)
    int oc = mtile * 16 + laneLo;           // this lane's out-channel (A row)

    v8f acc = {};
#pragma unroll
    for (int ry = 0; ry < 3; ++ry)
#pragma unroll
        for (int rx = 0; rx < 3; ++rx)
#pragma unroll
            for (int icb = 0; icb < HID; icb += 32) {
                const bf16_t* ap = wlds + oc * K2 + (ry * 3 + rx) * HID + icb;
                const bf16_t* bp = hlds + ((ry * 34 + lc + rx) << 6) + icb;
                v16bf af, bfr;
#pragma unroll
                for (int i = 0; i < 16; ++i) {
                    int fk = frag_k(i, kq);
                    af[i]  = ap[fk];
                    bfr[i] = bp[fk];        // contiguous 8-runs -> ds_load_b128
                }
                acc = __builtin_amdgcn_wmma_f32_16x16x32_bf16(
                          false, af, false, bfr, (short)0, acc, false, false);
            }
#pragma unroll
    for (int r = 0; r < 8; ++r) {
        int row = mtile * 16 + r + (kq << 3);
        float v = acc[r] + bias[row];
        h2[((size_t)(b * HID + row) << 16) + yrow * IMG + x0 + lc] =
            v > 0.f ? v : 0.f;
    }
}

// ---------------------------------------------------------------------
// conv3: 64 -> 1 channel, 3x3 SAME (M=1: no WMMA fit, small).
// ---------------------------------------------------------------------
__global__ void conv3_kernel(const float* __restrict__ h2,
                             const float* __restrict__ w,    // [1][64][3][3]
                             const float* __restrict__ bias, // [1]
                             float* __restrict__ tmp2) {
    int idx = blockIdx.x * blockDim.x + threadIdx.x;   // b*IMG2 + pix
    int pix = idx & (IMG2 - 1);
    int b   = idx >> 16;
    int y = pix >> 8, x = pix & 255;
    float s = bias[0];
    for (int ic = 0; ic < HID; ++ic) {
        const float* ch = h2 + ((size_t)(b * HID + ic) << 16);
        const float* wc = w + ic * 9;
#pragma unroll
        for (int ky = 0; ky < 3; ++ky)
#pragma unroll
            for (int kx = 0; kx < 3; ++kx) {
                int yy = y + ky - 1, xx = x + kx - 1;
                if (yy >= 0 && yy < IMG && xx >= 0 && xx < IMG)
                    s += wc[ky * 3 + kx] * ch[yy * IMG + xx];
            }
    }
    tmp2[idx] = s;
}

// ---------------------------------------------------------------------
// img2patch: [B,1,256,256] -> bf16 patch matrix [B][3969][64] (row-major)
// ---------------------------------------------------------------------
__global__ void img2patch_kernel(const float* __restrict__ in,
                                 bf16_t* __restrict__ patch) {
    int idx = blockIdx.x * blockDim.x + threadIdx.x;
    if (idx >= BATCH * NNODES * PP) return;
    int k = idx & 63;
    int t = idx >> 6;
    int n = t % NNODES;
    int b = t / NNODES;
    int pi = n / PHS, pj = n % PHS;
    int y = pi * 4 + (k >> 3);
    int x = pj * 4 + (k & 7);
    patch[idx] = (bf16_t)in[b * IMG2 + y * IMG + x];
}

// ---------------------------------------------------------------------
// Small weight GEMM:  out^T[b,:,m] = A[b,m,:] @ W   (A bf16 row-major,
// W f32 [K][F]).  W^T is staged once per block into LDS as bf16 so both
// fragments are K-contiguous b128 loads.  Output is written
// TRANSPOSED ([B][F][NNODES]) -- that is the layout the adjacency GEMM's
// TDM 2-D panel load wants.
// ---------------------------------------------------------------------
template <int K, int F>
__global__ void gemm_w_wmma_kernel(const bf16_t* __restrict__ A,   // [B][NNODES][K]
                                   const float* __restrict__ W,    // [K][F]
                                   bf16_t* __restrict__ outT) {    // [B][F][NNODES]
    extern __shared__ char dynsm[];
    bf16_t* wldsT = (bf16_t*)dynsm;               // [F][K]
    int tid = threadIdx.x;
    for (int t = tid; t < K * F; t += 256) {      // coalesced read, scatter write
        int k = t / F, f = t % F;
        wldsT[f * K + k] = (bf16_t)W[t];
    }
    __syncthreads();

    int rowTile = blockIdx.x * 8 + (tid >> 5);
    if (rowTile >= (NNODES + 15) / 16) return;    // wave-uniform, after barrier
    int lane   = tid & 31;
    int laneLo = lane & 15;
    int kq     = lane >> 4;
    int b      = blockIdx.y;
    int m      = rowTile * 16 + laneLo;
    if (m >= NNODES) m = NNODES - 1;              // clamp loads, guard stores
    const bf16_t* arow = A + ((size_t)b * NNODES + m) * K;

    v8f acc[F / 16] = {};
#pragma unroll
    for (int k0 = 0; k0 < K; k0 += 32) {
        v16bf af;
#pragma unroll
        for (int i = 0; i < 16; ++i) af[i] = arow[k0 + frag_k(i, kq)];
#pragma unroll
        for (int f = 0; f < F / 16; ++f) {
            const bf16_t* bp = wldsT + (f * 16 + laneLo) * K + k0;
            v16bf bfr;
#pragma unroll
            for (int i = 0; i < 16; ++i) bfr[i] = bp[frag_k(i, kq)];
            acc[f] = __builtin_amdgcn_wmma_f32_16x16x32_bf16(
                         false, af, false, bfr, (short)0, acc[f], false, false);
        }
    }
#pragma unroll
    for (int f = 0; f < F / 16; ++f)
#pragma unroll
        for (int r = 0; r < 8; ++r) {
            int row = rowTile * 16 + r + (kq << 3);
            int col = f * 16 + laneLo;
            if (row < NNODES)   // transposed store: [B][F][N]
                outT[((size_t)b * F + col) * NNODES + row] = (bf16_t)acc[f][r];
        }
}

// ---------------------------------------------------------------------
// Big adjacency GEMM:  C = adj[b] @ S + bias, S given transposed
// ST[b] = S^T : [F][NNODES] bf16.  K = 3969.
// adj is the HBM roofline term (252 MB/use): streamed fp32, converted to
// bf16 in registers.  The S K-panel (shared by all 8 waves) is staged by
// the Tensor Data Mover as a 2-D tile (KB contiguous K x F rows), landing
// in LDS as [F][KB] -- K-contiguous for b128 B-fragment loads.
// No early return (barriers): inactive row tiles clamp loads, guard stores.
// ---------------------------------------------------------------------
template <int F, bool RELU, bool OUTF32>
__global__ void gemm_adj_wmma_kernel(const float* __restrict__ adj,  // [B][N][N]
                                     const bf16_t* __restrict__ ST,  // [B][F][N]
                                     const float* __restrict__ bias, // [F]
                                     bf16_t* __restrict__ outb,      // [B][N][F]
                                     float* __restrict__ outf) {
    constexpr int KB = 128;                         // K panel (LDS: F*KB*2 bytes)
    extern __shared__ char dynsm[];
    bf16_t* sT = (bf16_t*)dynsm;                    // [F][KB]

    int tid    = threadIdx.x;
    int lane   = tid & 31, laneLo = lane & 15, kq = lane >> 4;
    int rowTile = blockIdx.x * 8 + (tid >> 5);
    bool active = rowTile < (NNODES + 15) / 16;
    int b      = blockIdx.y;
    int m      = rowTile * 16 + laneLo;
    if (m >= NNODES) m = NNODES - 1;
    const float*  arow = adj + ((size_t)b * NNODES + m) * NNODES;
    const bf16_t* Sb   = ST + (size_t)b * NNODES * F;

    v8f acc[F / 16] = {};
    constexpr int KMAIN = (NNODES / KB) * KB;       // 3968
    for (int k0 = 0; k0 < KMAIN; k0 += KB) {
        __syncthreads();                            // previous panel consumed
        if (tid == 0) {
            tdm_load_2d(0u, (const void*)(Sb + k0), KB, F, NNODES, F);
            __builtin_amdgcn_s_wait_tensorcnt(0);
        }
        __syncthreads();                            // panel visible to all waves
        __builtin_prefetch(arow + k0 + KB, 0, 0);   // -> global_prefetch_b8
#pragma unroll
        for (int ks = 0; ks < KB; ks += 32) {
            v16bf af;
#pragma unroll
            for (int i = 0; i < 16; ++i)
                af[i] = (bf16_t)arow[k0 + ks + frag_k(i, kq)];
#pragma unroll
            for (int f = 0; f < F / 16; ++f) {
                const bf16_t* bp = sT + (f * 16 + laneLo) * KB + ks;
                v16bf bfr;
#pragma unroll
                for (int i = 0; i < 16; ++i) bfr[i] = bp[frag_k(i, kq)];
                acc[f] = __builtin_amdgcn_wmma_f32_16x16x32_bf16(
                             false, af, false, bfr, (short)0, acc[f], false, false);
            }
        }
    }
    // K tail (3969 - 3968 = 1), straight from global ST, per-element guarded
    {
        int lim = NNODES - KMAIN;
        v16bf af;
#pragma unroll
        for (int i = 0; i < 16; ++i) {
            int kk = frag_k(i, kq);
            af[i] = (kk < lim) ? (bf16_t)arow[KMAIN + kk] : (bf16_t)0.f;
        }
#pragma unroll
        for (int f = 0; f < F / 16; ++f) {
            int col = f * 16 + laneLo;
            v16bf bfr;
#pragma unroll
            for (int i = 0; i < 16; ++i) {
                int kk = frag_k(i, kq);
                bfr[i] = (kk < lim) ? Sb[(size_t)col * NNODES + KMAIN + kk]
                                    : (bf16_t)0.f;
            }
            acc[f] = __builtin_amdgcn_wmma_f32_16x16x32_bf16(
                         false, af, false, bfr, (short)0, acc[f], false, false);
        }
    }
#pragma unroll
    for (int f = 0; f < F / 16; ++f)
#pragma unroll
        for (int r = 0; r < 8; ++r) {
            int row = rowTile * 16 + r + (kq << 3);
            int col = f * 16 + laneLo;
            float v = acc[f][r] + bias[col];
            if (RELU) v = v > 0.f ? v : 0.f;
            if (active && row < NNODES) {
                size_t o = ((size_t)b * NNODES + row) * F + col; // row-major out
                if (OUTF32) outf[o] = v;
                else        outb[o] = (bf16_t)v;
            }
        }
}

// ---------------------------------------------------------------------
// Final fuse: tmp1 (DC step) + tmp2 (CNN) + tmp3 (patch2img(g2)/mask), relu.
// patch2img done as a gather over the <=4 overlapping patches; mask == count.
// ---------------------------------------------------------------------
__global__ void combine_kernel(const float* __restrict__ in,
                               const float* __restrict__ proj,
                               const float* __restrict__ lamp,
                               const float* __restrict__ tmp2,
                               const float* __restrict__ g2,   // [B][N][64] f32
                               float* __restrict__ out) {
    int idx = blockIdx.x * blockDim.x + threadIdx.x;   // b*IMG2 + pix
    int pix = idx & (IMG2 - 1);
    int b   = idx >> 16;
    int y = pix >> 8, x = pix & 255;
    float lam = lamp[0];
    float a = in[idx];
    float t1 = a + lam * (proj[idx] - a);

    int piLo = (y - 4) >> 2; if (piLo < 0) piLo = 0;
    int piHi = y >> 2;       if (piHi > PHS - 1) piHi = PHS - 1;
    int pjLo = (x - 4) >> 2; if (pjLo < 0) pjLo = 0;
    int pjHi = x >> 2;       if (pjHi > PHS - 1) pjHi = PHS - 1;

    float sum = 0.f;
    int cnt = 0;
    for (int pi = piLo; pi <= piHi; ++pi)
        for (int pj = pjLo; pj <= pjHi; ++pj) {
            int n  = pi * PHS + pj;
            int kk = (y - pi * 4) * 8 + (x - pj * 4);
            sum += g2[((size_t)b * NNODES + n) * PP + kk];
            ++cnt;
        }
    float v = t1 + tmp2[idx] + sum / (float)cnt;
    out[idx] = v > 0.f ? v : 0.f;
}

// ---------------------------------------------------------------------
extern "C" void kernel_launch(void* const* d_in, const int* in_sizes, int n_in,
                              void* d_out, int out_size, void* d_ws, size_t ws_size,
                              hipStream_t stream) {
    (void)in_sizes; (void)n_in; (void)out_size; (void)ws_size;
    const float* in   = (const float*)d_in[0];
    const float* proj = (const float*)d_in[1];
    const float* adj  = (const float*)d_in[2];
    const float* lam  = (const float*)d_in[3];
    const float* w1   = (const float*)d_in[4];
    const float* b1   = (const float*)d_in[5];
    const float* w2   = (const float*)d_in[6];
    const float* b2   = (const float*)d_in[7];
    const float* w3c  = (const float*)d_in[8];
    const float* b3c  = (const float*)d_in[9];
    const float* gw3  = (const float*)d_in[10];
    const float* gb3  = (const float*)d_in[11];
    const float* gw4  = (const float*)d_in[12];
    const float* gb4  = (const float*)d_in[13];
    float* out = (float*)d_out;

    // workspace carve-up (256B aligned)
    char* base = (char*)d_ws;
    size_t o = 0;
    auto take = [&](size_t bytes) {
        char* p = base + o;
        o = (o + bytes + 255) & ~(size_t)255;
        return p;
    };
    float*  h1       = (float*)take((size_t)BATCH * HID * IMG2 * 4);   // 67 MB
    float*  h2       = (float*)take((size_t)BATCH * HID * IMG2 * 4);   // 67 MB
    float*  tmp2f    = (float*)take((size_t)BATCH * IMG2 * 4);         // 1 MB
    float*  g2       = (float*)take((size_t)BATCH * NNODES * PP * 4);  // 4 MB
    bf16_t* patch    = (bf16_t*)take((size_t)BATCH * NNODES * PP * 2); // 2 MB
    bf16_t* supportT = (bf16_t*)take((size_t)BATCH * NNODES * GH * 2); // 4 MB [B][F][N]
    bf16_t* g1       = (bf16_t*)take((size_t)BATCH * NNODES * GH * 2); // 4 MB [B][N][F]
    bf16_t* s2T      = (bf16_t*)take((size_t)BATCH * NNODES * PP * 2); // 2 MB [B][F][N]

    // ---- CNN branch ----
    conv1_relu_kernel<<<(BATCH * HID * IMG2) / 256, 256, 0, stream>>>(in, w1, b1, h1);
    conv2_relu_wmma_kernel<<<dim3(IMG2 / 32, BATCH), 256,
                             (HID * K2 + HID * 3 * 34) * sizeof(bf16_t), stream>>>(
        h1, w2, b2, h2);
    conv3_kernel<<<(BATCH * IMG2) / 256, 256, 0, stream>>>(h2, w3c, b3c, tmp2f);

    // ---- GCN branch ----
    img2patch_kernel<<<(BATCH * NNODES * PP + 255) / 256, 256, 0, stream>>>(in, patch);
    // support^T = (patch @ gcn_w3)^T
    gemm_w_wmma_kernel<PP, GH><<<dim3(32, BATCH), 256,
                                 PP * GH * sizeof(bf16_t), stream>>>(patch, gw3, supportT);
    // g1 = relu(adj @ support + gcn_b3)           (row-major out, feeds next A)
    gemm_adj_wmma_kernel<GH, true, false>
        <<<dim3(32, BATCH), 256, 128 * GH * sizeof(bf16_t), stream>>>(
            adj, supportT, gb3, g1, nullptr);
    // s2^T = (g1 @ gcn_w4)^T
    gemm_w_wmma_kernel<GH, PP><<<dim3(32, BATCH), 256,
                                 GH * PP * sizeof(bf16_t), stream>>>(g1, gw4, s2T);
    // g2 = adj @ s2 + gcn_b4
    gemm_adj_wmma_kernel<PP, false, true>
        <<<dim3(32, BATCH), 256, 128 * PP * sizeof(bf16_t), stream>>>(
            adj, s2T, gb4, nullptr, g2);

    // ---- fuse everything ----
    combine_kernel<<<(BATCH * IMG2) / 256, 256, 0, stream>>>(in, proj, lam, tmp2f, g2, out);
}
